// HamLSTMCell_79362405695492
// MI455X (gfx1250) — compile-verified, matrix operand order checked by reading, not run on hardware
//
#include <hip/hip_runtime.h>
#include <hip/hip_bf16.h>

// ---------------------------------------------------------------------------
// HAM-LSTM cell for MI455X (gfx1250, wave32, WMMA).
// All GEMM operands pre-staged as bf16 so the WMMA inner loop is pure
// global_load_b128 + v_wmma_f32_16x16x32_bf16 (fp32 accumulate).
// ---------------------------------------------------------------------------

typedef __attribute__((ext_vector_type(16))) __bf16 v16bf;
typedef __attribute__((ext_vector_type(8)))  float  v8f;
typedef __attribute__((ext_vector_type(4)))  float  v4f;
typedef __attribute__((ext_vector_type(4)))  unsigned int uint4v;

union V16 { v16bf v; unsigned int u[8]; };

__device__ __forceinline__ unsigned int f2bf(float f) {
    unsigned int x = __float_as_uint(f);
    unsigned int r = x + 0x7FFFu + ((x >> 16) & 1u);   // RNE to bf16
    return r >> 16;
}
__device__ __forceinline__ float bf2f(unsigned short s) {
    return __uint_as_float((unsigned int)s << 16);
}
__device__ __forceinline__ float sigmoidf(float x) {
    return 1.0f / (1.0f + __expf(-x));
}

#define WMMA_BF16(a, b, c) \
    __builtin_amdgcn_wmma_f32_16x16x32_bf16(false, (a), false, (b), (short)0, (c), false, false)

// ---- bf16 tile loader matching CDNA5 16-bit A/B WMMA VGPR layout ----------
// lane<16 : K = k0+0..7 (v0..3), K = k0+16..23 (v4..7)
// lane>=16: K = k0+8..15,        K = k0+24..31          (koff = 0 or 8)
__device__ __forceinline__ v16bf load_t16(const unsigned short* p, int koff) {
    const uint4v* q0 = (const uint4v*)(p + koff);
    const uint4v* q1 = (const uint4v*)(p + 16 + koff);
    uint4v a = *q0, b = *q1;
    V16 r;
#pragma unroll
    for (int i = 0; i < 4; ++i) { r.u[i] = a[i]; r.u[4 + i] = b[i]; }
    return r.v;
}

// ---- elementwise fp32 -> bf16 (8 elems/thread, vector loads/stores) -------
__global__ __launch_bounds__(256) void cvt_bf16(const float* __restrict__ in,
                                                unsigned short* __restrict__ out,
                                                int count8) {
    int idx = blockIdx.x * blockDim.x + threadIdx.x;
    if (idx >= count8) return;
    const v4f* p = (const v4f*)(in + (size_t)idx * 8);
    v4f f0 = p[0], f1 = p[1];
    uint4v o;
#pragma unroll
    for (int i = 0; i < 2; ++i) {
        o[i]     = f2bf(f0[2 * i]) | (f2bf(f0[2 * i + 1]) << 16);
        o[2 + i] = f2bf(f1[2 * i]) | (f2bf(f1[2 * i + 1]) << 16);
    }
    *(uint4v*)(out + (size_t)idx * 8) = o;
}

// ---- weight prep: fp32 [K,N] -> bf16 [N,K] (transposed, contiguous K) -----
__global__ __launch_bounds__(256) void prep_weights(const float* __restrict__ W,
                                                    unsigned short* __restrict__ Wt,
                                                    int K, int N) {
    int idx = blockIdx.x * blockDim.x + threadIdx.x;
    if (idx >= K * N) return;
    int n = idx / K, k = idx - n * K;
    Wt[idx] = (unsigned short)f2bf(W[(size_t)k * N + n]);
}

// ---- main WMMA GEMM: C[M,512] = act(A[M,K] @ W[K,512]) --------------------
// Wave tile 32x64 (2x4 WMMA tiles), block = 8 waves -> 128x128, grid.y = 4.
// CONCAT: A row r = [leafBf[r, 0:512], h0Bf[r>>8, 0:512]]   (K = 1024)
// EPI 0 : relu -> OutF32 and OutBf
// EPI 1 : relu -> OutBf only
// EPI 2 : cand = sigmoid; blend with upd/attn/leaf -> OutWrite (leaf_h_new)
template <bool CONCAT, int EPI>
__global__ __launch_bounds__(256) void wmma_gemm(
    const unsigned short* __restrict__ A, const unsigned short* __restrict__ ctrlBf,
    const unsigned short* __restrict__ Bt,
    float* __restrict__ OutF32, unsigned short* __restrict__ OutBf,
    const float* __restrict__ leaf, const float* __restrict__ attn,
    const float* __restrict__ upd, float* __restrict__ OutWrite,
    int M, int K) {
    const int lane = threadIdx.x & 31;
    const int wave = threadIdx.x >> 5;
    const int wm = wave >> 1;                          // 0..3
    const int wn = wave & 1;                           // 0..1
    const int rowBase = blockIdx.x * 128 + wm * 32;
    const int colBase = blockIdx.y * 128 + wn * 64;
    const int r    = lane & 15;
    const int koff = (lane < 16) ? 0 : 8;
    const int AS   = CONCAT ? 512 : K;                 // A row stride

    const int ra0 = min(rowBase + r,      M - 1);
    const int ra1 = min(rowBase + 16 + r, M - 1);

    const unsigned short* bp[4];
#pragma unroll
    for (int j = 0; j < 4; ++j) bp[j] = Bt + (size_t)(colBase + j * 16 + r) * K;

    v8f acc[2][4] = {};

    for (int k0 = 0; k0 < K; k0 += 32) {
        const unsigned short *arow0, *arow1;
        if (CONCAT && k0 >= 512) {
            arow0 = ctrlBf + (size_t)(ra0 >> 8) * 512 + (k0 - 512);
            arow1 = ctrlBf + (size_t)(ra1 >> 8) * 512 + (k0 - 512);
        } else {
            arow0 = A + (size_t)ra0 * AS + k0;
            arow1 = A + (size_t)ra1 * AS + k0;
        }
        v16bf a0 = load_t16(arow0, koff);
        v16bf a1 = load_t16(arow1, koff);
#pragma unroll
        for (int j = 0; j < 4; ++j) {
            v16bf b = load_t16(bp[j] + k0, koff);
            acc[0][j] = WMMA_BF16(a0, b, acc[0][j]);
            acc[1][j] = WMMA_BF16(a1, b, acc[1][j]);
        }
    }

    // C/D layout: col = lane&15; row = v + 8*(lane>>4)
    const int colb  = colBase + (lane & 15);
    const int rbase = 8 * (lane >> 4);
#pragma unroll
    for (int mt = 0; mt < 2; ++mt) {
#pragma unroll
        for (int j = 0; j < 4; ++j) {
#pragma unroll
            for (int v = 0; v < 8; ++v) {
                int row = rowBase + mt * 16 + rbase + v;
                int col = colb + j * 16;
                if (row < M) {
                    float val = acc[mt][j][v];
                    size_t idx = (size_t)row * 512 + col;
                    if (EPI == 0) {
                        float rl = fmaxf(val, 0.0f);
                        OutF32[idx] = rl;
                        OutBf[idx]  = (unsigned short)f2bf(rl);
                    } else if (EPI == 1) {
                        OutBf[idx] = (unsigned short)f2bf(fmaxf(val, 0.0f));
                    } else {
                        float cand = sigmoidf(val);
                        float u = upd[row];
                        float a = attn[row];
                        float lh = leaf[idx];
                        float w = u * cand + (1.0f - u) * lh;
                        OutWrite[idx] = a * w + (1.0f - a) * lh;
                    }
                }
            }
        }
    }
}

// ---- upd = sigmoid([leaf_h, ctrl] @ W_T), one wave per row ----------------
__global__ __launch_bounds__(256) void upd_kernel(const float* __restrict__ leaf,
                                                  const float* __restrict__ h0,
                                                  const float* __restrict__ WT,
                                                  float* __restrict__ upd) {
    int wave = threadIdx.x >> 5, lane = threadIdx.x & 31;
    int row = blockIdx.x * 8 + wave;       // 16384 rows
    int b = row >> 8;
    float acc = 0.0f;
    for (int k = lane; k < 512; k += 32) acc += leaf[(size_t)row * 512 + k] * WT[k];
    for (int k = lane; k < 512; k += 32) acc += h0[(size_t)b * 512 + k] * WT[512 + k];
#pragma unroll
    for (int m = 16; m; m >>= 1) acc += __shfl_xor(acc, m, 32);
    if (lane == 0) upd[row] = sigmoidf(acc);
}

// ---- tree search: per-batch search dots + top-down attn propagation -------
__global__ __launch_bounds__(256) void tree_kernel(const unsigned short* __restrict__ treeBf,
                                                   const float* __restrict__ h0,
                                                   const float* __restrict__ Ws,
                                                   float* __restrict__ attnOut) {
    int b = blockIdx.x, t = threadIdx.x;
    __shared__ float ws[1024];
    __shared__ float mr[256];
    __shared__ float at[512];
    __shared__ float red[256];
    ws[t] = Ws[t]; ws[256 + t] = Ws[256 + t];
    ws[512 + t] = Ws[512 + t]; ws[768 + t] = Ws[768 + t];
    __syncthreads();
    // control part of the search dot is shared by every node of this batch
    red[t] = h0[(size_t)b * 512 + t] * ws[512 + t]
           + h0[(size_t)b * 512 + 256 + t] * ws[768 + t];
    __syncthreads();
    for (int s = 128; s; s >>= 1) { if (t < s) red[t] += red[t + s]; __syncthreads(); }
    float cdot = red[0];

    if (t >= 1) {                       // internal heap nodes i = 1..255
        int i = t;
        int d = 31 - __clz(i);          // depth from root
        int lvl = 8 - d;                // reference level index (8 = root)
        int j = i - (1 << d);
        size_t off = 0;
        for (int l = 1; l < lvl; ++l) off += (size_t)64 * (256 >> l) * 512;
        const unsigned short* node = treeBf + off + ((size_t)b * (256 >> lvl) + j) * 512;
        float acc = cdot;
        for (int k = 0; k < 512; ++k) acc += bf2f(node[k]) * ws[k];
        mr[i] = sigmoidf(acc);
    }
    if (t == 0) at[1] = 1.0f;
    __syncthreads();
    for (int d = 1; d <= 8; ++d) {
        if (t < (1 << d)) {
            int i = (1 << d) + t, p = i >> 1;
            float m = mr[p];
            at[i] = (i & 1) ? at[p] * m : at[p] * (1.0f - m);
        }
        __syncthreads();
    }
    attnOut[(size_t)b * 256 + t] = at[256 + t];   // leaves = heap 256..511
}

// ---- tree_out[b,e] = sum_n attn[b,n] * inputs[b,n,e] ----------------------
__global__ __launch_bounds__(512) void treeout_kernel(const float* __restrict__ inputs,
                                                      const float* __restrict__ attn,
                                                      float* __restrict__ tout) {
    int b = blockIdx.x, e = threadIdx.x;
    __shared__ float a[256];
    if (e < 256) a[e] = attn[(size_t)b * 256 + e];
    __syncthreads();
    float acc = 0.0f;
    const float* ip = inputs + (size_t)b * 256 * 512 + e;
    for (int n = 0; n < 256; ++n) acc += a[n] * ip[(size_t)n * 512];
    tout[(size_t)b * 512 + e] = acc;
}

// ---- fused LSTM cell: z = x@Wk + h@Wr + bias; gates i,f,c,o ---------------
__global__ __launch_bounds__(512) void lstm_kernel(const float* __restrict__ tout,
                                                   const float* __restrict__ h0,
                                                   const float* __restrict__ c0,
                                                   const float* __restrict__ Wk,
                                                   const float* __restrict__ Wr,
                                                   const float* __restrict__ bias,
                                                   float* __restrict__ out) {
    int b = blockIdx.x, u = threadIdx.x;
    __shared__ float sx[512], sh[512];
    sx[u] = tout[(size_t)b * 512 + u];
    sh[u] = h0[(size_t)b * 512 + u];
    __syncthreads();
    float zi = bias[u], zf = bias[512 + u], zg = bias[1024 + u], zo = bias[1536 + u];
    for (int k = 0; k < 512; ++k) {
        float x = sx[k], h = sh[k];
        const float* kr = Wk + (size_t)k * 2048;
        const float* rr = Wr + (size_t)k * 2048;
        zi += x * kr[u]        + h * rr[u];
        zf += x * kr[512 + u]  + h * rr[512 + u];
        zg += x * kr[1024 + u] + h * rr[1024 + u];
        zo += x * kr[1536 + u] + h * rr[1536 + u];
    }
    float c = sigmoidf(zf) * c0[(size_t)b * 512 + u] + sigmoidf(zi) * tanhf(zg);
    float h = sigmoidf(zo) * tanhf(c);
    out[(size_t)b * 512 + u] = h;                 // h_new
    out[32768 + (size_t)b * 512 + u] = c;         // c_new
}

// ---------------------------------------------------------------------------
extern "C" void kernel_launch(void* const* d_in, const int* in_sizes, int n_in,
                              void* d_out, int out_size, void* d_ws, size_t ws_size,
                              hipStream_t stream) {
    const float* inputs = (const float*)d_in[0];   // [64,256,512]
    const float* h0     = (const float*)d_in[1];   // [64,512]
    const float* c0     = (const float*)d_in[2];   // [64,512]
    const float* W_t    = (const float*)d_in[3];   // [512,512]
    const float* W_j    = (const float*)d_in[4];   // [1024,512]
    const float* W_s    = (const float*)d_in[5];   // [1024,1]
    const float* W_h    = (const float*)d_in[6];   // [1024,512]
    const float* W_T    = (const float*)d_in[7];   // [1024,1]
    const float* Wk     = (const float*)d_in[8];   // [512,2048]
    const float* Wr     = (const float*)d_in[9];   // [512,2048]
    const float* bias   = (const float*)d_in[10];  // [2048]
    float* out = (float*)d_out;                    // h(32768) ++ c(32768) ++ leaf(8388608)

    // workspace layout
    float* ws   = (float*)d_ws;
    float* leaf = ws;                                   // 16384*512 fp32
    float* attn = leaf + (size_t)16384 * 512;           // 16384
    float* upd  = attn + 16384;                         // 16384
    float* tout = upd + 16384;                          // 32768
    unsigned short* inBf   = (unsigned short*)(tout + 32768);   // 16384*512
    unsigned short* leafBf = inBf + (size_t)16384 * 512;        // 16384*512
    unsigned short* treeBf = leafBf + (size_t)16384 * 512;      // 64*255*512
    unsigned short* h0Bf   = treeBf + (size_t)64 * 255 * 512;   // 32768
    unsigned short* wtT    = h0Bf + 32768;                      // 512*512
    unsigned short* wjT    = wtT + (size_t)512 * 512;           // 512*1024
    unsigned short* whT    = wjT + (size_t)512 * 1024;          // 512*1024

    // 1) stage bf16 operands: inputs, h0; weights transposed [N,K]
    cvt_bf16<<<(16384 * 512 / 8 + 255) / 256, 256, 0, stream>>>(inputs, inBf, 16384 * 512 / 8);
    cvt_bf16<<<(32768 / 8 + 255) / 256, 256, 0, stream>>>(h0, h0Bf, 32768 / 8);
    prep_weights<<<(512 * 512 + 255) / 256, 256, 0, stream>>>(W_t, wtT, 512, 512);
    prep_weights<<<(1024 * 512 + 255) / 256, 256, 0, stream>>>(W_j, wjT, 1024, 512);
    prep_weights<<<(1024 * 512 + 255) / 256, 256, 0, stream>>>(W_h, whT, 1024, 512);

    // 2) leaf_h = relu(inputs @ W_t): M=16384, K=512 -> fp32 + bf16
    wmma_gemm<false, 0><<<dim3(128, 4), 256, 0, stream>>>(
        inBf, nullptr, wtT, leaf, leafBf, nullptr, nullptr, nullptr, nullptr, 16384, 512);

    // 3) bottom-up join: level l has M = 64*(256>>l) rows, K=1024 -> bf16
    size_t off[9]; off[1] = 0;
    for (int l = 2; l <= 8; ++l) off[l] = off[l - 1] + (size_t)64 * (256 >> (l - 1)) * 512;
    for (int l = 1; l <= 8; ++l) {
        int M = 64 * (256 >> l);
        const unsigned short* A = (l == 1) ? leafBf : treeBf + off[l - 1];  // [M,1024] view
        wmma_gemm<false, 1><<<dim3((M + 127) / 128, 4), 256, 0, stream>>>(
            A, nullptr, wjT, nullptr, treeBf + off[l], nullptr, nullptr, nullptr, nullptr, M, 1024);
    }

    // 4) top-down search -> leaf attentions
    tree_kernel<<<64, 256, 0, stream>>>(treeBf, h0, W_s, attn);

    // 5) tree_out = sum_n attn * inputs
    treeout_kernel<<<64, 512, 0, stream>>>(inputs, attn, tout);

    // 6) upd = sigmoid([leaf_h, ctrl] @ W_T)
    upd_kernel<<<2048, 256, 0, stream>>>(leaf, h0, W_T, upd);

    // 7) cand GEMM + fused write blend -> leaf_h_new (d_out + 65536)
    wmma_gemm<true, 2><<<dim3(128, 4), 256, 0, stream>>>(
        leafBf, h0Bf, whT, nullptr, nullptr, leaf, attn, upd, out + 65536, 16384, 1024);

    // 8) LSTM cell -> h_new, c_new
    lstm_kernel<<<64, 512, 0, stream>>>(tout, h0, c0, Wk, Wr, bias, out);
}
